// SPGloss_49100066128562
// MI455X (gfx1250) — matrix operand order repaired and checked by reading, not compiled, not run on hardware
//
#include <hip/hip_runtime.h>
#include <hip/hip_bf16.h>

typedef __attribute__((ext_vector_type(16))) _Float16 v16h;
typedef __attribute__((ext_vector_type(8)))  float    v8f;

#define IMG_B   64
#define IMG_HW  65536
#define NSB     320          // 5 scales * 64 batches
#define ALPHA0f 0.1667f
#define ALPHA1f 0.8333f
#define CAP     4096

__device__ __forceinline__ float softplusf(float x) {
  float ax = fabsf(x);
  return fmaxf(x, 0.f) + log1pf(__expf(-ax));
}

// ---------------------------------------------------------------------------
// K0: zero hist / candidate counters / output scalar
// ---------------------------------------------------------------------------
__global__ __launch_bounds__(256) void zero_kernel(unsigned* __restrict__ hist,
                                                   unsigned* __restrict__ ccnt,
                                                   float* __restrict__ out) {
  int i = blockIdx.x * 256 + threadIdx.x;
  if (i < NSB * 256) hist[i] = 0u;
  if (i < NSB)       ccnt[i] = 0u;
  if (i == 0)        out[0]  = 0.f;
}

// ---------------------------------------------------------------------------
// K1: ring weights via WMMA box filters.
// One wave per 16x16 tile. Box sum = banded-Toeplitz matmul:
//   vertical:  D = A_k(16x32) x I(32x16)   (two halves -> 32 haloed columns)
//   horizontal:S = T(16x32)  x B_k(32x16)
// f16 operands are exact (integer counts <= 25). Result packed as rw + 2*tp.
// ---------------------------------------------------------------------------
__global__ __launch_bounds__(256) void rings_kernel(const int* __restrict__ tgt,
                                                    float* __restrict__ rings) {
  __shared__ _Float16 sI[8][32][32];   // haloed target tile per wave
  __shared__ _Float16 sT[8][16][32];   // vertical-sum intermediate per wave

  const int lane = threadIdx.x & 31;
  const int wv   = threadIdx.x >> 5;
  const int tile = blockIdx.x * 8 + wv;          // 16384 tiles total
  const int b    = tile >> 8;
  const int tt   = tile & 255;
  const int R0   = (tt >> 4) << 4;
  const int C0   = (tt & 15) << 4;
  const int n    = lane & 15;                    // N / M-within-half index
  const int hh   = lane >> 4;                    // lane half

  // Stage haloed 20x32 region (rows j<20 valid; rest zero -> annihilated by band)
  for (int e = lane; e < 1024; e += 32) {
    int j = e >> 5, c = e & 31;
    int r = R0 - 2 + j, gc = C0 - 2 + c;
    float tv = 0.f;
    if (j < 20 && r >= 0 && r < 256 && gc >= 0 && gc < 256)
      tv = (float)tgt[((size_t)b * 256 + r) * 256 + gc];
    sI[wv][j][c] = (_Float16)tv;
  }
  __syncthreads();

  // B fragments of the input (32x16 f16): VGPR v holds rows K=h*16+2v,+1; N=lane%16
  v16h bl, br;
  for (int v = 0; v < 8; ++v) {
    int k0 = hh * 16 + 2 * v;
    bl[2 * v]     = sI[wv][k0][n];
    bl[2 * v + 1] = sI[wv][k0 + 1][n];
    br[2 * v]     = sI[wv][k0][n + 16];
    br[2 * v + 1] = sI[wv][k0 + 1][n + 16];
  }

  auto boxsum = [&](int lo, int kk) -> v8f {
    // A band (16x32): row m = lane%16; ones at K in [m+2-lo, m+2-lo+kk-1]
    v16h aV;
    for (int v = 0; v < 8; ++v) {
      int g = v >> 2, vv = v & 3;
      int kb = hh * 16 + v * 2;
      int j0 = n + 2 - lo, j1 = j0 + kk - 1;
      aV[2 * v]     = (_Float16)((kb     >= j0 && kb     <= j1) ? 1.f : 0.f);
      aV[2 * v + 1] = (_Float16)((kb + 1 >= j0 && kb + 1 <= j1) ? 1.f : 0.f);
    }
    v8f z = {};
    v8f dl = __builtin_amdgcn_wmma_f32_16x16x32_f16(false, aV, false, bl, (short)0, z, false, false);
    v8f dr = __builtin_amdgcn_wmma_f32_16x16x32_f16(false, aV, false, br, (short)0, z, false, false);
    // D layout: VGPR v -> M = v + 8*hh, N = n. Stage T (16 rows x 32 haloed cols).
    for (int v = 0; v < 8; ++v) {
      sT[wv][v + 8 * hh][n]      = (_Float16)dl[v];
      sT[wv][v + 8 * hh][n + 16] = (_Float16)dr[v];
    }
    __syncthreads();
    // A fragment of T: row m = n, K pairs per A-layout
    v16h aT;
    for (int v = 0; v < 8; ++v) {
      int g = v >> 2, vv = v & 3;
      int kb = hh * 16 + v * 2;
      aT[2 * v]     = sT[wv][n][kb];
      aT[2 * v + 1] = sT[wv][n][kb + 1];
    }
    // B band (32x16): col c = n; ones at K in [c+2-lo, c+2-lo+kk-1]
    v16h bB;
    for (int v = 0; v < 8; ++v) {
      int k0 = hh * 16 + 2 * v;
      int j0 = n + 2 - lo, j1 = j0 + kk - 1;
      bB[2 * v]     = (_Float16)((k0     >= j0 && k0     <= j1) ? 1.f : 0.f);
      bB[2 * v + 1] = (_Float16)((k0 + 1 >= j0 && k0 + 1 <= j1) ? 1.f : 0.f);
    }
    v8f S = __builtin_amdgcn_wmma_f32_16x16x32_f16(false, aT, false, bB, (short)0, z, false, false);
    __syncthreads();
    return S;
  };

  v8f S5 = boxsum(2, 5);   // conv5 'same': pad (2,2)
  v8f S4 = boxsum(1, 4);   // conv4: pad (1,2)
  v8f S3 = boxsum(1, 3);   // conv3: pad (1,1)
  v8f S2 = boxsum(0, 2);   // conv2: pad (0,1)

  for (int v = 0; v < 8; ++v) {
    float tp = (S5[v] > 0.f) ? 1.f : 0.f;
    float r1 = ((tp - S4[v]) > 0.f) ? 0.5f : 0.f;
    float r2 = ((S4[v] - S3[v]) > 0.f) ? 0.3f : 0.f;
    float r3 = ((S3[v] - S2[v]) > 0.f) ? 0.2f : 0.f;
    int row = R0 + v + 8 * hh, col = C0 + n;
    rings[(size_t)b * IMG_HW + row * 256 + col] = (r1 + r2 + r3) + 2.f * tp;
  }
}

// ---------------------------------------------------------------------------
// K2: fused base loss + 256-bin float-key histogram of loss_p (single read of
// all logits). base = A1*tgt*sp(-x) + A0*(1-tgt)*rw*sp(x); loss_p = lw*(1-tp).
// ---------------------------------------------------------------------------
__global__ __launch_bounds__(256) void loss_kernel(
    const float* __restrict__ l0, const float* __restrict__ l1,
    const float* __restrict__ l2, const float* __restrict__ l3,
    const float* __restrict__ l4,
    const int* __restrict__ tgt, const float* __restrict__ rings,
    unsigned* __restrict__ hist, float* __restrict__ out) {
  int bid = blockIdx.x;
  int chunk = bid & 15;
  int b = (bid >> 4) & 63;
  int s = bid >> 10;
  const float* xs = s == 0 ? l0 : s == 1 ? l1 : s == 2 ? l2 : s == 3 ? l3 : l4;

  __shared__ unsigned lh[256];
  __shared__ float red[256];
  lh[threadIdx.x] = 0u;
  __syncthreads();

  float base = 0.f;
  int p0 = chunk * 4096;
  for (int i = threadIdx.x; i < 4096; i += 256) {
    size_t off = (size_t)b * IMG_HW + p0 + i;
    float x  = xs[off];
    float tg = (float)tgt[off];
    float rv = rings[off];
    float tp = (rv >= 2.f) ? 1.f : 0.f;
    float rw = rv - 2.f * tp;
    float spx = softplusf(x), spnx = softplusf(-x);
    base += ALPHA1f * tg * spnx + ALPHA0f * (1.f - tg) * rw * spx;
    float lp = (spx - x * tg) * (1.f - tp);       // >= 0 -> bits are monotonic
    atomicAdd(&lh[__float_as_uint(lp) >> 24], 1u);
  }
  __syncthreads();
  int sb = s * 64 + b;
  if (lh[threadIdx.x]) atomicAdd(&hist[sb * 256 + threadIdx.x], lh[threadIdx.x]);

  red[threadIdx.x] = base;
  __syncthreads();
  for (int st = 128; st > 0; st >>= 1) {
    if ((int)threadIdx.x < st) red[threadIdx.x] += red[threadIdx.x + st];
    __syncthreads();
  }
  if (threadIdx.x == 0) atomicAdd(out, red[0]);
}

// ---------------------------------------------------------------------------
// K3: per (scale,batch) threshold bin such that count(bin >= t) >= 200
// ---------------------------------------------------------------------------
__global__ void thresh_kernel(const unsigned* __restrict__ hist,
                              unsigned* __restrict__ thr) {
  int sb = blockIdx.x * blockDim.x + threadIdx.x;
  if (sb >= NSB) return;
  const unsigned* h = &hist[sb * 256];
  unsigned cum = 0; int t = 0;
  for (int bin = 255; bin >= 0; --bin) {
    cum += h[bin];
    if (cum >= 200) { t = bin; break; }
  }
  thr[sb] = (unsigned)t;
}

// ---------------------------------------------------------------------------
// K4: compact candidates (value,index) with key-bin >= threshold
// ---------------------------------------------------------------------------
__global__ __launch_bounds__(256) void compact_kernel(
    const float* __restrict__ l0, const float* __restrict__ l1,
    const float* __restrict__ l2, const float* __restrict__ l3,
    const float* __restrict__ l4,
    const int* __restrict__ tgt, const float* __restrict__ rings,
    const unsigned* __restrict__ thr, unsigned* __restrict__ ccnt,
    float* __restrict__ cval, int* __restrict__ cidx) {
  int bid = blockIdx.x;
  int chunk = bid & 15;
  int b = (bid >> 4) & 63;
  int s = bid >> 10;
  const float* xs = s == 0 ? l0 : s == 1 ? l1 : s == 2 ? l2 : s == 3 ? l3 : l4;
  int sb = s * 64 + b;
  unsigned tb = thr[sb];
  int p0 = chunk * 4096;
  for (int i = threadIdx.x; i < 4096; i += 256) {
    int p = p0 + i;
    size_t off = (size_t)b * IMG_HW + p;
    float x  = xs[off];
    float tg = (float)tgt[off];
    float rv = rings[off];
    float tp = (rv >= 2.f) ? 1.f : 0.f;
    float lp = (softplusf(x) - x * tg) * (1.f - tp);
    if ((__float_as_uint(lp) >> 24) >= tb) {
      unsigned pos = atomicAdd(&ccnt[sb], 1u);
      if (pos < CAP) {
        cval[(size_t)sb * CAP + pos] = lp;
        cidx[(size_t)sb * CAP + pos] = p;
      }
    }
  }
}

// ---------------------------------------------------------------------------
// K5: per (scale,batch): bitonic sort candidates desc (ties: lower idx first),
// pick hard ranks perm[:60]+20 (perm shared across batch, per scale) + 60
// per-batch random easy indices, dedupe (set semantics), add deltas.
// delta(selected pixel) = A0 * (1 - tp) * softplus(x).
// ---------------------------------------------------------------------------
__global__ __launch_bounds__(256) void select_kernel(
    const float* __restrict__ l0, const float* __restrict__ l1,
    const float* __restrict__ l2, const float* __restrict__ l3,
    const float* __restrict__ l4,
    const float* __restrict__ rings,
    const float* __restrict__ cval, const int* __restrict__ cidx,
    const unsigned* __restrict__ ccnt, float* __restrict__ out) {
  __shared__ float sv[CAP];
  __shared__ int   si[CAP];
  __shared__ int   sel[120];
  __shared__ int   perm[130];
  __shared__ float red[256];

  int sb = blockIdx.x;
  int s = sb >> 6, b = sb & 63;
  int nc = (int)ccnt[sb]; if (nc > CAP) nc = CAP;

  for (int i = threadIdx.x; i < CAP; i += 256) {
    if (i < nc) { sv[i] = cval[(size_t)sb * CAP + i]; si[i] = cidx[(size_t)sb * CAP + i]; }
    else        { sv[i] = -1.f; si[i] = 0x7FFFFFFF; }
  }
  __syncthreads();

  // bitonic sort, descending by (value, then index ascending)
  for (int k = 2; k <= CAP; k <<= 1) {
    for (int j = k >> 1; j > 0; j >>= 1) {
      for (int i = threadIdx.x; i < CAP; i += 256) {
        int p = i ^ j;
        if (p > i) {
          bool desc = ((i & k) == 0);
          float vi = sv[i], vp = sv[p];
          int ii = si[i], ip = si[p];
          bool iless = (vi < vp) || (vi == vp && ii > ip);
          if (desc ? iless : !iless) {
            sv[i] = vp; sv[p] = vi; si[i] = ip; si[p] = ii;
          }
        }
      }
      __syncthreads();
    }
  }

  if (threadIdx.x == 0) {
    // hard: Fisher-Yates perm of 130 (seeded per scale), ranks perm[:60]+20
    unsigned st = 0x243F6A88u ^ (unsigned)(s * 0x9E3779B1u);
    for (int i = 0; i < 130; ++i) perm[i] = i;
    for (int i = 129; i > 0; --i) {
      st = st * 1664525u + 1013904223u;
      int j = (int)(st % (unsigned)(i + 1));
      int tmp = perm[i]; perm[i] = perm[j]; perm[j] = tmp;
    }
    for (int i = 0; i < 60; ++i) {
      int r = 20 + perm[i];
      sel[i] = (r < CAP) ? si[r] : 0x7FFFFFFF;
    }
    // easy: 60 per-batch pseudo-random pixel indices
    unsigned se = 0x85EBCA6Bu ^ (unsigned)(s * 0xC2B2AE35u) ^ (unsigned)(b * 0x27D4EB2Fu);
    for (int i = 0; i < 60; ++i) {
      se = se * 1664525u + 1013904223u;
      sel[60 + i] = (int)((se >> 8) & 65535);
    }
  }
  __syncthreads();

  float acc = 0.f;
  if (threadIdx.x < 120) {
    int p = sel[threadIdx.x];
    bool dup = false;
    for (int j = 0; j < (int)threadIdx.x; ++j) if (sel[j] == p) dup = true;
    if (!dup && p >= 0 && p < IMG_HW) {
      const float* xs = s == 0 ? l0 : s == 1 ? l1 : s == 2 ? l2 : s == 3 ? l3 : l4;
      size_t off = (size_t)b * IMG_HW + p;
      float x  = xs[off];
      float rv = rings[off];
      float tp = (rv >= 2.f) ? 1.f : 0.f;
      acc = (1.f - tp) * ALPHA0f * softplusf(x);
    }
  }
  red[threadIdx.x] = acc;
  __syncthreads();
  for (int st2 = 128; st2 > 0; st2 >>= 1) {
    if ((int)threadIdx.x < st2) red[threadIdx.x] += red[threadIdx.x + st2];
    __syncthreads();
  }
  if (threadIdx.x == 0) atomicAdd(out, red[0]);
}

// ---------------------------------------------------------------------------
extern "C" void kernel_launch(void* const* d_in, const int* in_sizes, int n_in,
                              void* d_out, int out_size, void* d_ws, size_t ws_size,
                              hipStream_t stream) {
  (void)in_sizes; (void)n_in; (void)out_size; (void)ws_size;
  const float* l0 = (const float*)d_in[0];
  const float* l1 = (const float*)d_in[1];
  const float* l2 = (const float*)d_in[2];
  const float* l3 = (const float*)d_in[3];
  const float* l4 = (const float*)d_in[4];
  const int*  tgt = (const int*)d_in[5];
  float* out = (float*)d_out;

  char* ws = (char*)d_ws;
  const size_t SZ_RINGS = (size_t)IMG_B * IMG_HW * sizeof(float);   // 16 MB
  const size_t SZ_HIST  = (size_t)NSB * 256 * sizeof(unsigned);     // 320 KB
  const size_t SZ_THR   = (size_t)NSB * sizeof(unsigned);
  const size_t SZ_CCNT  = (size_t)NSB * sizeof(unsigned);
  const size_t SZ_CVAL  = (size_t)NSB * CAP * sizeof(float);        // 5.2 MB

  float*    rings = (float*)ws;
  unsigned* hist  = (unsigned*)(ws + SZ_RINGS);
  unsigned* thr   = (unsigned*)(ws + SZ_RINGS + SZ_HIST);
  unsigned* ccnt  = (unsigned*)(ws + SZ_RINGS + SZ_HIST + SZ_THR);
  float*    cval  = (float*)(ws + SZ_RINGS + SZ_HIST + SZ_THR + SZ_CCNT);
  int*      cidx  = (int*)(ws + SZ_RINGS + SZ_HIST + SZ_THR + SZ_CCNT + SZ_CVAL);

  zero_kernel   <<<NSB, 256, 0, stream>>>(hist, ccnt, out);
  rings_kernel  <<<2048, 256, 0, stream>>>(tgt, rings);
  loss_kernel   <<<5120, 256, 0, stream>>>(l0, l1, l2, l3, l4, tgt, rings, hist, out);
  thresh_kernel <<<5, 64, 0, stream>>>(hist, thr);
  compact_kernel<<<5120, 256, 0, stream>>>(l0, l1, l2, l3, l4, tgt, rings, thr, ccnt, cval, cidx);
  select_kernel <<<NSB, 256, 0, stream>>>(l0, l1, l2, l3, l4, rings, cval, cidx, ccnt, out);
}
